// GCastEdgeEncoder_9191230013921
// MI455X (gfx1250) — compile-verified, matrix operand order checked by reading, not run on hardware
//
#include <hip/hip_runtime.h>
#include <hip/hip_bf16.h>

typedef __attribute__((ext_vector_type(16))) __bf16 v16bf;
typedef __attribute__((ext_vector_type(8)))  __bf16 v8bf;
typedef __attribute__((ext_vector_type(4)))  __bf16 v4bf;
typedef __attribute__((ext_vector_type(8)))  float  v8f;

#define D_FEAT 128
#define IN_CH  384        // 3 * D_FEAT
#define OUT_CH 512
#define M_TILE 32         // edges per workgroup
#define XPAD   392        // bf16 per x-row in LDS (384 + 8) -> rows stay 16B aligned
#define ZPAD   516        // f32 per z-row in LDS (512 + 4)  -> rows stay 16B aligned

__device__ __forceinline__ float fast_sigmoid(float x) {
    // 1 / (1 + e^-x) with single v_rcp_f32 (no IEEE-div expansion)
    return __builtin_amdgcn_rcpf(1.0f + __expf(-x));
}

// ---------------------------------------------------------------------------
// Kernel 1: convert W (512x384 f32, row-major [out, in]) to bf16 in workspace.
// ---------------------------------------------------------------------------
__global__ void conv_w_bf16(const float* __restrict__ W,
                            unsigned short* __restrict__ Wb, int n4)
{
    int i = blockIdx.x * blockDim.x + threadIdx.x;
    if (i < n4) {
        const float4 v = ((const float4*)W)[i];
        v4bf o;
        o.x = (__bf16)v.x; o.y = (__bf16)v.y; o.z = (__bf16)v.z; o.w = (__bf16)v.w;
        ((v4bf*)Wb)[i] = o;
    }
}

// ---------------------------------------------------------------------------
// Kernel 2: fused gather -> bf16 GEMM (WMMA) -> bias -> silu(silu) -> LayerNorm
// 256 threads = 8 waves. Workgroup tile: 32 edges x 512 outputs.
// Wave w owns columns [w*64, w*64+64) as 4 WMMA N-tiles; 2 WMMA M-tiles.
// ---------------------------------------------------------------------------
__global__ __launch_bounds__(256) void edge_mlp_ln(
    const float* __restrict__ sender,
    const float* __restrict__ receiver,
    const int*   __restrict__ eidx,      // [2, E]: row 0 senders, row 1 receivers
    const float* __restrict__ eattr,
    const unsigned short* __restrict__ Wb16,
    const float* __restrict__ bias,
    const float* __restrict__ gvec,
    const float* __restrict__ bvec,
    float* __restrict__ out,
    int nE)
{
    __shared__ __align__(16) unsigned char smem[M_TILE * ZPAD * 4]; // 66048 B
    __shared__ float sum_buf[M_TILE * 8];
    __shared__ float sq_buf [M_TILE * 8];
    __bf16* xl = (__bf16*)smem;   // phase 1: 32 x XPAD bf16 (25088 B)
    float*  zl = (float*) smem;   // phase 2: 32 x ZPAD f32  (66048 B)

    const int tid  = threadIdx.x;
    const int wave = tid >> 5;
    const int lane = tid & 31;
    const int h    = lane >> 4;   // half-wave (WMMA operand striping)
    const int nl   = lane & 15;
    const int e0   = blockIdx.x * M_TILE;

    // ---- gather + f32->bf16 into LDS: 32 rows x 96 float4 = 3072 float4 ----
    #pragma unroll
    for (int i = 0; i < 12; ++i) {
        int idx4 = tid + i * 256;           // 0..3071
        int row  = idx4 / 96;
        int col  = (idx4 % 96) * 4;
        int e    = e0 + row; if (e >= nE) e = nE - 1;
        const float* src;
        if (col < D_FEAT)
            src = sender   + (size_t)eidx[e]      * D_FEAT + col;
        else if (col < 2 * D_FEAT)
            src = receiver + (size_t)eidx[nE + e] * D_FEAT + (col - D_FEAT);
        else
            src = eattr    + (size_t)e            * D_FEAT + (col - 2 * D_FEAT);
        float4 v = *(const float4*)src;
        v4bf o;
        o.x = (__bf16)v.x; o.y = (__bf16)v.y; o.z = (__bf16)v.z; o.w = (__bf16)v.w;
        *(v4bf*)&xl[row * XPAD + col] = o;
    }
    __syncthreads();

    // ---- WMMA main loop: K = 384 in 12 steps of 32 ----
    const __bf16* Wb = reinterpret_cast<const __bf16*>(Wb16);
    v8f acc[2][4];
    #pragma unroll
    for (int mt = 0; mt < 2; ++mt)
        #pragma unroll
        for (int t = 0; t < 4; ++t)
            acc[mt][t] = (v8f)0.0f;

    for (int kb = 0; kb < 12; ++kb) {
        const int kbase = kb * 32;
        // B fragments: B[k][n] = W[n][k]; lane (nl,h) holds col n = tile*16+nl,
        // elems 0..7 = K kbase+8h.., elems 8..15 = K kbase+16+8h..
        v16bf bfrag[4];
        #pragma unroll
        for (int t = 0; t < 4; ++t) {
            const __bf16* wr = Wb + (size_t)((wave * 4 + t) * 16 + nl) * IN_CH
                                  + kbase + 8 * h;
            v8bf lo = *(const v8bf*)(wr);
            v8bf hi = *(const v8bf*)(wr + 16);
            #pragma unroll
            for (int j = 0; j < 8; ++j) { bfrag[t][j] = lo[j]; bfrag[t][j + 8] = hi[j]; }
        }
        // A fragments from LDS (same K striping, row m = mtile*16 + nl)
        v16bf afrag[2];
        #pragma unroll
        for (int mt = 0; mt < 2; ++mt) {
            const __bf16* xr = xl + (mt * 16 + nl) * XPAD + kbase + 8 * h;
            v8bf lo = *(const v8bf*)(xr);
            v8bf hi = *(const v8bf*)(xr + 16);
            #pragma unroll
            for (int j = 0; j < 8; ++j) { afrag[mt][j] = lo[j]; afrag[mt][j + 8] = hi[j]; }
        }
        #pragma unroll
        for (int mt = 0; mt < 2; ++mt)
            #pragma unroll
            for (int t = 0; t < 4; ++t)
                acc[mt][t] = __builtin_amdgcn_wmma_f32_16x16x32_bf16(
                    false, afrag[mt], false, bfrag[t],
                    (short)0, acc[mt][t], false, false);
    }
    __syncthreads();   // x region dead; reuse LDS for z

    // ---- epilogue: bias + silu(silu(z)), spill fp32 tile to LDS ----
    // acc element r of tile (mt,t): row = mt*16 + 8h + r, col = (wave*4+t)*16 + nl
    #pragma unroll
    for (int t = 0; t < 4; ++t) {
        const int ncol = (wave * 4 + t) * 16 + nl;
        const float bn = bias[ncol];
        #pragma unroll
        for (int mt = 0; mt < 2; ++mt) {
            #pragma unroll
            for (int r = 0; r < 8; ++r) {
                const int row = mt * 16 + 8 * h + r;
                float z  = acc[mt][t][r] + bn;
                float s1 = z  * fast_sigmoid(z);    // silu
                float s2 = s1 * fast_sigmoid(s1);   // silu again
                zl[row * ZPAD + ncol] = s2;
            }
        }
    }
    __syncthreads();

    // ---- LayerNorm: 8 threads per row over 512 cols ----
    const int row = tid >> 3;
    const int sub = tid & 7;
    float s = 0.0f, q = 0.0f;
    #pragma unroll
    for (int c = sub * 64; c < sub * 64 + 64; c += 4) {
        float4 v = *(const float4*)&zl[row * ZPAD + c];
        s += v.x + v.y + v.z + v.w;
        q += v.x * v.x + v.y * v.y + v.z * v.z + v.w * v.w;
    }
    sum_buf[row * 8 + sub] = s;
    sq_buf [row * 8 + sub] = q;
    __syncthreads();

    float S = 0.0f, Q = 0.0f;
    #pragma unroll
    for (int i = 0; i < 8; ++i) { S += sum_buf[row * 8 + i]; Q += sq_buf[row * 8 + i]; }
    const float mean = S * (1.0f / 512.0f);
    const float var  = Q * (1.0f / 512.0f) - mean * mean;
    const float rstd = __builtin_amdgcn_rsqf(var + 1e-5f);

    const int e = e0 + row;
    if (e < nE) {
        float* orow = out + (size_t)e * OUT_CH;
        #pragma unroll
        for (int c = sub * 64; c < sub * 64 + 64; c += 4) {
            float4 v = *(const float4*)&zl[row * ZPAD + c];
            float4 g = *(const float4*)&gvec[c];
            float4 b = *(const float4*)&bvec[c];
            float4 o;
            o.x = (v.x - mean) * rstd * g.x + b.x;
            o.y = (v.y - mean) * rstd * g.y + b.y;
            o.z = (v.z - mean) * rstd * g.z + b.z;
            o.w = (v.w - mean) * rstd * g.w + b.w;
            *(float4*)&orow[c] = o;
        }
    }
}

// ---------------------------------------------------------------------------
extern "C" void kernel_launch(void* const* d_in, const int* in_sizes, int n_in,
                              void* d_out, int out_size, void* d_ws, size_t ws_size,
                              hipStream_t stream) {
    const float* sender   = (const float*)d_in[0];
    const float* receiver = (const float*)d_in[1];
    const int*   eidx     = (const int*)  d_in[2];
    const float* eattr    = (const float*)d_in[3];
    const float* W        = (const float*)d_in[4];
    const float* bias     = (const float*)d_in[5];
    const float* gamma    = (const float*)d_in[6];
    const float* beta     = (const float*)d_in[7];
    float* out            = (float*)d_out;
    unsigned short* Wb    = (unsigned short*)d_ws;   // 512*384 bf16 = 384 KB

    const int nE = in_sizes[2] / 2;                  // edge_index is [2, E]
    const int n4 = (OUT_CH * IN_CH) / 4;             // 49152 float4
    conv_w_bf16<<<(n4 + 255) / 256, 256, 0, stream>>>(W, Wb, n4);

    const int nblk = (nE + M_TILE - 1) / M_TILE;     // 15625 for E=500k
    edge_mlp_ln<<<nblk, 256, 0, stream>>>(sender, receiver, eidx, eattr,
                                          Wb, bias, gamma, beta, out, nE);
}
